// PatternBasedV2_69346541961310
// MI455X (gfx1250) — matrix operand order repaired and checked by reading, not compiled, not run on hardware
//
#include <hip/hip_runtime.h>

typedef _Float16 v16h __attribute__((ext_vector_type(16)));
typedef _Float16 h8   __attribute__((ext_vector_type(8)));
typedef _Float16 h4   __attribute__((ext_vector_type(4)));
typedef float    v8f  __attribute__((ext_vector_type(8)));

#define BTOT 16384
#define TB   4            // boards per block in kernel A (rows = TB*8 = 32 = 2 M-tiles)
#define NF   32
#define NM   128
#define EPSV 1e-5f

__device__ __constant__ int PAT_CELLS[84] = {
    0, 1, 2, 3, 4, 5, 6, 7, 9, 14,
    8, 9, 10, 11, 12, 13, 14, 15,
    16, 17, 18, 19, 20, 21, 22, 23,
    24, 25, 26, 27, 28, 29, 30, 31,
    0, 1, 2, 3, 4, 8, 9, 10, 11, 12,
    0, 1, 2, 3, 8, 9, 10, 16, 17, 24,
    3, 4, 10, 11, 17, 18, 24, 25, 32,
    5, 12, 19, 26, 33, 40,
    6, 13, 20, 27, 34, 41, 48,
    7, 20, 21, 28, 35, 42, 49, 56};
__device__ __constant__ int PAT_OFF[11] = {0, 10, 18, 26, 34, 44, 54, 63, 69, 76, 84};

struct WfPtrs { const float* p[10]; };

union H2 { _Float16 h[2]; unsigned int u; };

// ---- fragment helpers ------------------------------------------------------
__device__ __forceinline__ v16h ld16h(const _Float16* p) {   // 16 consecutive f16
    h8 lo = *(const h8*)p;
    h8 hh = *(const h8*)(p + 8);
    v16h r;
    #pragma unroll
    for (int e = 0; e < 8; ++e) { r[e] = lo[e]; r[e + 8] = hh[e]; }
    return r;
}
// A fragment from an LDS tile [rows][32] f16 with fused ReLU.
__device__ __forceinline__ v16h ldA_relu(const _Float16* t, int row, int hi) {
    h8 lo = *(const h8*)&t[row * NF + 16 * hi];
    h8 hh = *(const h8*)&t[row * NF + 16 * hi + 8];
    v16h r;
    #pragma unroll
    for (int e = 0; e < 8; ++e) {
        _Float16 u = lo[e]; r[e]     = (u > (_Float16)0.f) ? u : (_Float16)0.f;
        _Float16 w = hh[e]; r[e + 8] = (w > (_Float16)0.f) ? w : (_Float16)0.f;
    }
    return r;
}
__device__ __forceinline__ v8f bcast8(float b) {
    v8f c;
    #pragma unroll
    for (int v = 0; v < 8; ++v) c[v] = b;
    return c;
}
#define WMMA(a, b, c) __builtin_amdgcn_wmma_f32_16x16x32_f16(false, (a), false, (b), (short)0, (c), false, false)

// ---------------------------------------------------------------------------
// Prep 1: generic fp32 -> f16 weight conversion
// ---------------------------------------------------------------------------
__global__ __launch_bounds__(256) void k_cvt16(const float* __restrict__ src,
                                               _Float16* __restrict__ dst, int nelem)
{
    int i = blockIdx.x * 256 + threadIdx.x;
    if (i < nelem) dst[i] = (_Float16)src[i];
}

// ---------------------------------------------------------------------------
// Prep 2: pattern-embed weights padded to K=32 (f16) + gather-offset table
// ---------------------------------------------------------------------------
__global__ __launch_bounds__(256) void k_prep_embed(WfPtrs wf,
                                                    _Float16* __restrict__ wf16,
                                                    int* __restrict__ tab)
{
    const int t = blockIdx.x * 256 + threadIdx.x;
    if (t < 10 * 32 * 32) {
        const int g = t >> 10, o = (t >> 5) & 31, K = t & 31;
        const int L = PAT_OFF[g + 1] - PAT_OFF[g];
        wf16[t] = (K < 2 * L) ? (_Float16)wf.p[g][o * 2 * L + K] : (_Float16)0.f;
    }
    if (t < 10 * 8 * 32) {
        const int g = t >> 8, s = (t >> 5) & 7, K = t & 31;
        const int L = PAT_OFF[g + 1] - PAT_OFF[g];
        int val = -1;
        if (K < 2 * L) {
            const int plane = (K >= L) ? 1 : 0;
            const int cell  = PAT_CELLS[PAT_OFF[g] + (K - plane * L)];
            int ii = cell >> 3, jj = cell & 7;
            if (s & 2) ii = 7 - ii;                         // flip axis 3
            if (s & 4) jj = 7 - jj;                         // flip axis 4
            const int orig = (s & 1) ? (jj * 8 + ii) : (ii * 8 + jj);  // transpose
            val = plane * 64 + orig;
        }
        tab[t] = val;
    }
}

// ---------------------------------------------------------------------------
// Kernel A: D4 expansion + pattern embed + 3 residual stacks of grouped convs
// + last conv (32->128) + group/sym reduction  ->  b5 [BTOT][128] fp32
// One wave per pattern group; per-group activations in wave-private LDS tiles.
// N-tile pair uses logical channel o = 2n+nt so D stores pack to b32
// (backend merges them into ds_store_2addr_b32).
// ---------------------------------------------------------------------------
__global__ __launch_bounds__(320) void k_embed_conv(
    const float* __restrict__ x,
    const _Float16* __restrict__ wf16, const int* __restrict__ tab,
    const _Float16* __restrict__ cw16, const float* __restrict__ conv_b,
    const _Float16* __restrict__ lw16, const float* __restrict__ last_b,
    float* __restrict__ b5)
{
    __shared__ __align__(16) float    xs[TB * 128];          // 2 KB  staged input
    __shared__ __align__(16) _Float16 hbuf[10][32 * NF];     // 20 KB residual h
    __shared__ __align__(16) _Float16 tbuf[10][32 * NF];     // 20 KB conv temp t
    __shared__ __align__(16) float    pbuf[10][TB * NM];     // 20 KB b5 partials

    const int tid = threadIdx.x;
    const int blk = blockIdx.x;

    if (tid < 128)
        ((float4*)xs)[tid] = ((const float4*)(x + (size_t)blk * (TB * 128)))[tid];
    __syncthreads();

    const int lane = tid & 31;
    const int n    = lane & 15;   // N-column / A-row within 16
    const int hi   = lane >> 4;   // lane half
    const int g    = tid >> 5;    // wave = pattern group 0..9

    _Float16* hb = hbuf[g];
    _Float16* tb = tbuf[g];

    // ---------------- pattern embedding (table gather + WMMA) ---------------
    {
        v16h a[2];
        #pragma unroll
        for (int mt = 0; mt < 2; ++mt) {
            const int row = mt * 16 + n;                    // row = board*8 + sym
            const float* xb = &xs[(row >> 3) * 128];
            const int*   tg = tab + (g * 8 + (row & 7)) * 32;
            #pragma unroll
            for (int e = 0; e < 16; ++e) {
                const int i = e >> 1, h = e & 1;
                const int K = 16 * hi + e;  // A layout
                const int t = tg[K];                        // branch-free gather
                const float v = xb[t < 0 ? 0 : t];
                a[mt][e] = (t < 0) ? (_Float16)0.f : (_Float16)v;
            }
        }
        const _Float16* wg = wf16 + g * 32 * NF;
        v16h b0 = ld16h(wg + (2 * n) * NF + 16 * hi);       // logical ch 2n
        v16h b1 = ld16h(wg + (2 * n + 1) * NF + 16 * hi);   // logical ch 2n+1
        v8f c00 = {}, c01 = {}, c10 = {}, c11 = {};
        c00 = WMMA(a[0], b0, c00); c01 = WMMA(a[0], b1, c01);
        c10 = WMMA(a[1], b0, c10); c11 = WMMA(a[1], b1, c11);
        const int r0 = 8 * hi;
        #pragma unroll
        for (int v2 = 0; v2 < 8; ++v2) {                    // packed b32 stores
            H2 p0; p0.h[0] = (_Float16)c00[v2]; p0.h[1] = (_Float16)c01[v2];
            H2 p1; p1.h[0] = (_Float16)c10[v2]; p1.h[1] = (_Float16)c11[v2];
            *(unsigned int*)&hb[(r0 + v2) * NF + 2 * n]      = p0.u;
            *(unsigned int*)&hb[(16 + r0 + v2) * NF + 2 * n] = p1.u;
        }
    }

    // ---------------- 3 residual stacks x 3 grouped 1x1 convs ---------------
    for (int st = 0; st < 3; ++st) {
        for (int ly = 0; ly < 3; ++ly) {
            const int layer = st * 3 + ly;
            const _Float16* src = (ly == 0) ? hb : tb;
            const _Float16* wl  = cw16 + (size_t)(layer * 10 + g) * NF * NF;
            const float*    bv  = conv_b + (size_t)(layer * 10 + g) * NF;
            v16h a0 = ldA_relu(src, n, hi);
            v16h a1 = ldA_relu(src, 16 + n, hi);
            v16h b0 = ld16h(wl + (2 * n) * NF + 16 * hi);
            v16h b1 = ld16h(wl + (2 * n + 1) * NF + 16 * hi);
            // bias rides in the WMMA C operand (D = A*B + C)
            v8f c00 = bcast8(bv[2 * n]),     c01 = bcast8(bv[2 * n + 1]);
            v8f c10 = c00,                   c11 = c01;
            c00 = WMMA(a0, b0, c00); c01 = WMMA(a0, b1, c01);
            c10 = WMMA(a1, b0, c10); c11 = WMMA(a1, b1, c11);
            const int r0 = 8 * hi;
            #pragma unroll
            for (int v2 = 0; v2 < 8; ++v2) {
                H2 p0; p0.h[0] = (_Float16)c00[v2]; p0.h[1] = (_Float16)c01[v2];
                H2 p1; p1.h[0] = (_Float16)c10[v2]; p1.h[1] = (_Float16)c11[v2];
                *(unsigned int*)&tb[(r0 + v2) * NF + 2 * n]      = p0.u;
                *(unsigned int*)&tb[(16 + r0 + v2) * NF + 2 * n] = p1.u;
            }
        }
        // residual h += t, vectorized to b128 (packed f16 adds)
        {
            h8* hb8 = (h8*)hb;
            h8* tb8 = (h8*)tb;
            #pragma unroll
            for (int it = 0; it < 4; ++it) {
                const int i2 = lane + it * 32;
                h8 av = hb8[i2], bv8 = tb8[i2];
                #pragma unroll
                for (int e = 0; e < 8; ++e) av[e] = av[e] + bv8[e];
                hb8[i2] = av;
            }
        }
    }

    // ---------------- last conv 32->128, +bias, sum over 8 syms -------------
    {
        const _Float16* wl = lw16 + (size_t)g * NM * NF;
        const float*    bv = last_b + (size_t)g * NM;
        v16h a0 = ldA_relu(hb, n, hi);
        v16h a1 = ldA_relu(hb, 16 + n, hi);
        for (int nt = 0; nt < 8; ++nt) {
            const int o = nt * 16 + n;
            v16h b = ld16h(wl + o * NF + 16 * hi);
            v8f c0 = bcast8(bv[o]);       // sum over v gives the 8x bias term
            v8f c1 = c0;
            c0 = WMMA(a0, b, c0);
            c1 = WMMA(a1, b, c1);
            // D VGPR index v spans the 8 syms of one board -> free sym-sum
            float s0 = 0.f, s1 = 0.f;
            #pragma unroll
            for (int v2 = 0; v2 < 8; ++v2) { s0 += c0[v2]; s1 += c1[v2]; }
            pbuf[g][hi * NM + o]       = s0;   // board = 0*2+hi
            pbuf[g][(2 + hi) * NM + o] = s1;   // board = 1*2+hi
        }
    }
    __syncthreads();
    if (tid < 128) {                                        // reduce over groups
        float4 s = make_float4(0.f, 0.f, 0.f, 0.f);
        #pragma unroll
        for (int w2 = 0; w2 < 10; ++w2) {
            float4 v = ((const float4*)pbuf[w2])[tid];
            s.x += v.x; s.y += v.y; s.z += v.z; s.w += v.w;
        }
        ((float4*)(b5 + (size_t)blk * (TB * NM)))[tid] = s;
    }
}

// ---------------------------------------------------------------------------
// Column mean/var over the 16384-row batch: deterministic two-pass reduction.
// ---------------------------------------------------------------------------
__global__ __launch_bounds__(128) void k_stats_part(const float* __restrict__ in,
                                                    float* __restrict__ part)
{
    const int c = threadIdx.x, b = blockIdx.x;
    const float* p = in + (size_t)b * 128 * NM + c;
    float s = 0.f, q = 0.f;
    for (int r = 0; r < 128; ++r) { float v = p[(size_t)r * NM]; s += v; q += v * v; }
    part[b * 256 + c] = s;
    part[b * 256 + 128 + c] = q;
}

__global__ __launch_bounds__(128) void k_stats_red(const float* __restrict__ part,
                                                   const float* __restrict__ gam,
                                                   const float* __restrict__ bet,
                                                   float* __restrict__ ss)
{
    const int c = threadIdx.x;
    float s = 0.f, q = 0.f;
    for (int i = 0; i < 128; ++i) { s += part[i * 256 + c]; q += part[i * 256 + 128 + c]; }
    const float inv  = 1.f / (float)BTOT;
    const float mean = s * inv;
    const float var  = q * inv - mean * mean;
    const float scale = gam[c] * rsqrtf(var + EPSV);
    ss[c]       = scale;
    ss[128 + c] = bet[c] - mean * scale;
}

// ---------------------------------------------------------------------------
// out = relu(in*scale+shift) @ W^T   ([16384x128] x [128x128]) via WMMA f16.
// ---------------------------------------------------------------------------
__global__ __launch_bounds__(256) void k_bn_gemm(const float* __restrict__ in,
                                                 const float* __restrict__ ss,
                                                 const _Float16* __restrict__ w16,
                                                 float* __restrict__ out)
{
    __shared__ __align__(16) _Float16 a16[128 * NM];   // 32 KB
    __shared__ float sc[NM], sh[NM];
    const int tid = threadIdx.x;
    if (tid < NM) { sc[tid] = ss[tid]; sh[tid] = ss[NM + tid]; }
    __syncthreads();

    const int blk = blockIdx.x;
    const float* inb = in + (size_t)blk * 128 * NM;
    {   // fused BN-affine + ReLU staging; each thread owns 4 fixed columns
        const int col0 = (tid * 4) & (NM - 1);
        const float s0 = sc[col0],     s1 = sc[col0 + 1];
        const float s2 = sc[col0 + 2], s3 = sc[col0 + 3];
        const float t0 = sh[col0],     t1 = sh[col0 + 1];
        const float t2 = sh[col0 + 2], t3 = sh[col0 + 3];
        #pragma unroll
        for (int k = 0; k < 16; ++k) {
            const int idx = tid * 4 + k * 1024;
            float4 v = *(const float4*)&inb[idx];
            float r0 = v.x * s0 + t0, r1 = v.y * s1 + t1;
            float r2 = v.z * s2 + t2, r3 = v.w * s3 + t3;
            h4 o;
            o[0] = (_Float16)(r0 > 0.f ? r0 : 0.f);
            o[1] = (_Float16)(r1 > 0.f ? r1 : 0.f);
            o[2] = (_Float16)(r2 > 0.f ? r2 : 0.f);
            o[3] = (_Float16)(r3 > 0.f ? r3 : 0.f);
            *(h4*)&a16[idx] = o;
        }
    }
    __syncthreads();

    const int mt = tid >> 5, lane = tid & 31, n = lane & 15, hi = lane >> 4;
    const int rowA = mt * 16 + n;                       // 8 waves = 8 M-tiles
    v16h a[4];                                          // A frags hoisted over nt
    #pragma unroll
    for (int ks = 0; ks < 4; ++ks) {                    // layout: K=16hi.. / 16hi+8..
        h8 clo = *(const h8*)&a16[rowA * NM + ks * 32 + 16 * hi];
        h8 chi = *(const h8*)&a16[rowA * NM + ks * 32 + 16 * hi + 8];
        #pragma unroll
        for (int e = 0; e < 8; ++e) { a[ks][e] = clo[e]; a[ks][e + 8] = chi[e]; }
    }

    for (int q = 0; q < 4; ++q) {                       // paired output columns
        const int o0 = q * 32 + 2 * n;                  // logical col for tile0
        v8f c0 = {}, c1 = {};
        #pragma unroll
        for (int ks = 0; ks < 4; ++ks) {                // K = 128 in 4 WMMA steps
            v16h b0 = ld16h(w16 + (size_t)o0 * NM + ks * 32 + 16 * hi);
            v16h b1 = ld16h(w16 + (size_t)(o0 + 1) * NM + ks * 32 + 16 * hi);
            c0 = WMMA(a[ks], b0, c0);
            c1 = WMMA(a[ks], b1, c1);
        }
        const int row0 = blk * 128 + mt * 16 + 8 * hi;
        #pragma unroll
        for (int v2 = 0; v2 < 8; ++v2) {                // adjacent cols -> b64
            float2 pr = make_float2(c0[v2], c1[v2]);
            *(float2*)&out[(size_t)(row0 + v2) * NM + o0] = pr;
        }
    }
}

// ---------------------------------------------------------------------------
// out[row] = relu(bn(h2)) . lin3_w + lin3_b
// ---------------------------------------------------------------------------
__global__ __launch_bounds__(256) void k_final(const float* __restrict__ in,
                                               const float* __restrict__ ss,
                                               const float* __restrict__ w3,
                                               const float* __restrict__ b3,
                                               float* __restrict__ out)
{
    __shared__ float sc[NM], sh[NM], ww[NM];
    const int tid = threadIdx.x;
    if (tid < NM) { sc[tid] = ss[tid]; sh[tid] = ss[NM + tid]; ww[tid] = w3[tid]; }
    __syncthreads();
    const int row = blockIdx.x * 256 + tid;
    const float* p = in + (size_t)row * NM;
    float acc = b3[0];
    #pragma unroll 4
    for (int c = 0; c < NM; c += 4) {
        float4 v = *(const float4*)&p[c];
        float r0 = v.x * sc[c] + sh[c];
        float r1 = v.y * sc[c + 1] + sh[c + 1];
        float r2 = v.z * sc[c + 2] + sh[c + 2];
        float r3 = v.w * sc[c + 3] + sh[c + 3];
        acc += (r0 > 0.f ? r0 : 0.f) * ww[c];
        acc += (r1 > 0.f ? r1 : 0.f) * ww[c + 1];
        acc += (r2 > 0.f ? r2 : 0.f) * ww[c + 2];
        acc += (r3 > 0.f ? r3 : 0.f) * ww[c + 3];
    }
    out[row] = acc;
}

// ---------------------------------------------------------------------------
extern "C" void kernel_launch(void* const* d_in, const int* in_sizes, int n_in,
                              void* d_out, int out_size, void* d_ws, size_t ws_size,
                              hipStream_t stream)
{
    (void)in_sizes; (void)n_in; (void)out_size; (void)ws_size;
    const float* x = (const float*)d_in[0];
    WfPtrs wf;
    for (int i = 0; i < 10; ++i) wf.p[i] = (const float*)d_in[1 + i];
    const float* conv_w = (const float*)d_in[11];
    const float* conv_b = (const float*)d_in[12];
    const float* last_w = (const float*)d_in[13];
    const float* last_b = (const float*)d_in[14];
    const float* bn_g   = (const float*)d_in[15];
    const float* bn_b   = (const float*)d_in[16];
    const float* lin1   = (const float*)d_in[17];
    const float* lin2   = (const float*)d_in[18];
    const float* lin3   = (const float*)d_in[19];
    const float* lin3b  = (const float*)d_in[20];

    // ---- workspace layout ----
    char* ws = (char*)d_ws;
    const size_t big = (size_t)BTOT * NM * sizeof(float);   // 8 MB
    float*    bufA = (float*)ws;                            // b5, later h2
    float*    bufB = (float*)(ws + big);                    // h1
    char*     sm   = ws + 2 * big;
    float*    part = (float*)sm;                 sm += 128 * 256 * 4;
    float*    ss   = (float*)sm;                 sm += 256 * 4;
    _Float16* cw16 = (_Float16*)sm;              sm += 9 * 10 * 32 * 32 * 2;
    _Float16* lw16 = (_Float16*)sm;              sm += 10 * 128 * 32 * 2;
    _Float16* l116 = (_Float16*)sm;              sm += 128 * 128 * 2;
    _Float16* l216 = (_Float16*)sm;              sm += 128 * 128 * 2;
    _Float16* wf16 = (_Float16*)sm;              sm += 10 * 32 * 32 * 2;
    int*      tab  = (int*)sm;                   sm += 10 * 8 * 32 * 4;

    // ---- weight prep (tiny; graph-captured like everything else) ----
    k_cvt16<<<(9 * 10 * 1024 + 255) / 256, 256, 0, stream>>>(conv_w, cw16, 9 * 10 * 1024);
    k_cvt16<<<(10 * 128 * 32 + 255) / 256, 256, 0, stream>>>(last_w, lw16, 10 * 128 * 32);
    k_cvt16<<<(128 * 128 + 255) / 256, 256, 0, stream>>>(lin1, l116, 128 * 128);
    k_cvt16<<<(128 * 128 + 255) / 256, 256, 0, stream>>>(lin2, l216, 128 * 128);
    k_prep_embed<<<40, 256, 0, stream>>>(wf, wf16, tab);

    // ---- heavy fused kernel ----
    k_embed_conv<<<BTOT / TB, 320, 0, stream>>>(x, wf16, tab, cw16, conv_b,
                                                lw16, last_b, bufA);

    // ---- BN / MLP chain (batch stats force kernel boundaries) ----
    k_stats_part<<<128, 128, 0, stream>>>(bufA, part);
    k_stats_red<<<1, 128, 0, stream>>>(part, bn_g + 0, bn_b + 0, ss);
    k_bn_gemm<<<128, 256, 0, stream>>>(bufA, ss, l116, bufB);

    k_stats_part<<<128, 128, 0, stream>>>(bufB, part);
    k_stats_red<<<1, 128, 0, stream>>>(part, bn_g + 128, bn_b + 128, ss);
    k_bn_gemm<<<128, 256, 0, stream>>>(bufB, ss, l216, bufA);

    k_stats_part<<<128, 128, 0, stream>>>(bufA, part);
    k_stats_red<<<1, 128, 0, stream>>>(part, bn_g + 256, bn_b + 256, ss);
    k_final<<<BTOT / 256, 256, 0, stream>>>(bufA, ss, lin3, lin3b, (float*)d_out);
}
